// Mamba2Layer_29446295781720
// MI455X (gfx1250) — matrix operand order repaired and checked by reading, not compiled, not run on hardware
//
#include <hip/hip_runtime.h>
#include <math.h>

#define D_MODEL 2048
#define D_STATE 64
#define D_CONV  4
#define D_INNER 4096
#define N_HEADS 64
#define HEAD_DIM 64
#define CHUNK 256
#define BATCH 2
#define SEQ 2048
#define NCHUNK (SEQ / CHUNK)
#define DP (2 * D_INNER + 2 * N_HEADS * D_STATE + N_HEADS) /* 16448 */
#define DP_PAD 16512                                       /* 129 * 128 */
#define OFF_XC (D_INNER)
#define OFF_B (2 * D_INNER)
#define OFF_C (2 * D_INNER + N_HEADS * D_STATE)
#define OFF_DT (2 * D_INNER + 2 * N_HEADS * D_STATE)
#define BT (BATCH * SEQ)
#define SSD_SCALE 0.125f
#define RMS_EPS 1.1920929e-07f

#if defined(__has_builtin)
#if __has_builtin(__builtin_amdgcn_global_load_async_to_lds_b128)
#define HAVE_ASYNC 1
#else
#define HAVE_ASYNC 0
#endif
#else
#define HAVE_ASYNC 0
#endif

#define GLOBAL_AS __attribute__((address_space(1)))
#define LDS_AS __attribute__((address_space(3)))

typedef __attribute__((__vector_size__(16))) int i32x4;
typedef __attribute__((ext_vector_type(16))) __bf16 v16bf;
typedef __attribute__((ext_vector_type(8)))  __bf16 v8bf;
typedef __attribute__((ext_vector_type(8)))  float  v8f;

static __device__ inline __bf16 f2bf(float f) {
  union { float f; unsigned u; } v; v.f = f;
  unsigned r = v.u + 0x7FFFu + ((v.u >> 16) & 1u);
  unsigned short h = (unsigned short)(r >> 16);
  __bf16 b; __builtin_memcpy(&b, &h, 2); return b;
}

static __device__ inline v8f wmma_bf16(v16bf a, v16bf b, v8f c) {
  return __builtin_amdgcn_wmma_f32_16x16x32_bf16(false, a, false, b, (short)0, c,
                                                 false, false);
}

static __device__ inline v8f zero8() {
  v8f z = {0.f, 0.f, 0.f, 0.f, 0.f, 0.f, 0.f, 0.f};
  return z;
}

static __device__ inline v16bf cat8(v8bf lo, v8bf hi) {
  return __builtin_shufflevector(lo, hi, 0, 1, 2, 3, 4, 5, 6, 7, 8, 9, 10, 11,
                                 12, 13, 14, 15);
}

// A-operand fragment (16 rows x K=32) from row-major f32 [rows][stride].
static __device__ inline v16bf afrag_g(const float* base, int stride, int row0,
                                       int k0, int lane) {
  const float* p =
      base + (size_t)(row0 + (lane & 15)) * stride + k0 + ((lane >> 4) * 16);
  v16bf f;
#pragma unroll
  for (int i = 0; i < 8; ++i) f[i] = f2bf(p[i]);
#pragma unroll
  for (int i = 0; i < 8; ++i) f[8 + i] = f2bf(p[8 + i]);
  return f;
}

// B-operand fragment from N x K row-major f32 (B^T storage): 16 contiguous K.
static __device__ inline v16bf bfrag_g(const float* base, int stride, int col0,
                                       int k0, int lane) {
  const float* p =
      base + (size_t)(col0 + (lane & 15)) * stride + k0 + ((lane >> 4) * 16);
  v16bf f;
#pragma unroll
  for (int i = 0; i < 16; ++i) f[i] = f2bf(p[i]);
  return f;
}

static __device__ inline v16bf afrag_lds(const __bf16* base, int stride,
                                         int row0, int k0, int lane) {
  const __bf16* p = base + (row0 + (lane & 15)) * stride + k0 + ((lane >> 4) * 16);
  v8bf lo = *(const v8bf*)p;
  v8bf hi = *(const v8bf*)(p + 8);
  return cat8(lo, hi);
}

static __device__ inline v16bf bfrag_lds(const __bf16* base, int stride,
                                         int col0, int k0, int lane) {
  const __bf16* p =
      base + (col0 + (lane & 15)) * stride + k0 + ((lane >> 4) * 16);
  v8bf lo = *(const v8bf*)p;
  v8bf hi = *(const v8bf*)(p + 8);
  return cat8(lo, hi);
}

// ---------------------------------------------------------------------------
// C[m,n] = sum_k A[m,k] * B[n,k]; A: M x K bf16 rm, B: N x K bf16 rm,
// C: f32, row stride LDC. All dims compile-time; M,N,K multiples of 128/128/32.
// Double-buffered LDS staging; async global->LDS copies when available.
// ---------------------------------------------------------------------------
template <int N, int K, int LDC>
__global__ __launch_bounds__(256) void gemm_bf16_wmma(
    const __bf16* __restrict__ A, const __bf16* __restrict__ B,
    float* __restrict__ C) {
  __shared__ __bf16 sA[2][128][40];
  __shared__ __bf16 sB[2][128][40];
  int tid = threadIdx.x;
  int lane = tid & 31, wv = tid >> 5;
  int wm = (wv >> 1) * 32;
  int wn = (wv & 1) * 64;
  int bm = blockIdx.y * 128, bn = blockIdx.x * 128;

  auto stage = [&](int buf, int k0) {
#pragma unroll
    for (int it = 0; it < 2; ++it) {
      int idx = tid + it * 256;
      int r = idx >> 2, cc = (idx & 3) * 8; // 8 bf16 = 16B chunk
      const __bf16* ga = A + (size_t)(bm + r) * K + k0 + cc;
      const __bf16* gb = B + (size_t)(bn + r) * K + k0 + cc;
#if HAVE_ASYNC
      __builtin_amdgcn_global_load_async_to_lds_b128(
          (GLOBAL_AS i32x4*)(ga), (LDS_AS i32x4*)(&sA[buf][r][cc]), 0, 0);
      __builtin_amdgcn_global_load_async_to_lds_b128(
          (GLOBAL_AS i32x4*)(gb), (LDS_AS i32x4*)(&sB[buf][r][cc]), 0, 0);
#else
      *(uint4*)(&sA[buf][r][cc]) = *(const uint4*)(ga);
      *(uint4*)(&sB[buf][r][cc]) = *(const uint4*)(gb);
#endif
    }
  };

  v8f acc[2][4];
#pragma unroll
  for (int i = 0; i < 2; ++i)
#pragma unroll
    for (int j = 0; j < 4; ++j) acc[i][j] = zero8();

  stage(0, 0);
  const int nk = K / 32;
  for (int ks = 0; ks < nk; ++ks) {
    int buf = ks & 1;
#if HAVE_ASYNC
#if __has_builtin(__builtin_amdgcn_s_wait_asynccnt)
    __builtin_amdgcn_s_wait_asynccnt(0);
#else
    asm volatile("s_wait_asynccnt 0" ::: "memory");
#endif
#endif
    __syncthreads();
    if (ks + 1 < nk) stage(buf ^ 1, (ks + 1) * 32);

    v16bf af[2], bf[4];
#pragma unroll
    for (int mt = 0; mt < 2; ++mt)
      af[mt] = afrag_lds(&sA[buf][0][0], 40, wm + mt * 16, 0, lane);
#pragma unroll
    for (int nt = 0; nt < 4; ++nt)
      bf[nt] = bfrag_lds(&sB[buf][0][0], 40, wn + nt * 16, 0, lane);
#pragma unroll
    for (int mt = 0; mt < 2; ++mt)
#pragma unroll
      for (int nt = 0; nt < 4; ++nt)
        acc[mt][nt] = wmma_bf16(af[mt], bf[nt], acc[mt][nt]);
  }

  int lo16 = lane & 15, hi8 = (lane >> 4) * 8;
  float* Cw = C + (size_t)(bm + wm + hi8) * LDC + bn + wn + lo16;
#pragma unroll
  for (int mt = 0; mt < 2; ++mt)
#pragma unroll
    for (int nt = 0; nt < 4; ++nt)
#pragma unroll
      for (int r = 0; r < 8; ++r)
        Cw[(size_t)(mt * 16 + r) * LDC + nt * 16] = acc[mt][nt][r];
}

// ---------------------------------------------------------------------------
// f32 -> bf16 conversions (done once; halves GEMM operand traffic).
// ---------------------------------------------------------------------------
__global__ __launch_bounds__(256) void cvt_f32_bf16(const float* __restrict__ s,
                                                    __bf16* __restrict__ d) {
  size_t gid = (size_t)blockIdx.x * 256 + threadIdx.x;
  d[gid] = f2bf(s[gid]);
}

// in_proj_w (16448 x 2048) -> bf16 padded to 16512 rows (pad = clamp row).
__global__ __launch_bounds__(256) void cvt_pad_w(const float* __restrict__ s,
                                                 __bf16* __restrict__ d) {
  size_t gid = (size_t)blockIdx.x * 256 + threadIdx.x;
  int k = (int)(gid & (D_MODEL - 1));
  size_t row = gid >> 11;
  size_t sr = row < DP ? row : (size_t)(DP - 1);
  d[gid] = f2bf(s[sr * D_MODEL + k]);
}

// ---------------------------------------------------------------------------
// Depthwise causal conv (width 4) + SiLU on the x_conv slice of proj.
// ---------------------------------------------------------------------------
__global__ __launch_bounds__(256) void conv_silu(const float* __restrict__ proj,
                                                 const float* __restrict__ cw,
                                                 float* __restrict__ xcbuf) {
  size_t gid = (size_t)blockIdx.x * 256 + threadIdx.x;
  int ch = (int)(gid & (D_INNER - 1));
  size_t rest = gid >> 12;
  int t = (int)(rest & (SEQ - 1));
  int b = (int)(rest >> 11);
  float a = 0.f;
#pragma unroll
  for (int k = 0; k < D_CONV; ++k) {
    int tt = t - (D_CONV - 1) + k;
    if (tt >= 0)
      a += cw[ch * D_CONV + k] *
           proj[(size_t)(b * SEQ + tt) * DP_PAD + OFF_XC + ch];
  }
  xcbuf[gid] = a / (1.f + __expf(-a));
}

// ---------------------------------------------------------------------------
// dt = clip(softplus(dt_raw + bias), 1e-4, 0.5); cum = cumsum(A*dt) per chunk.
// ---------------------------------------------------------------------------
__global__ __launch_bounds__(256) void dt_cumsum(const float* __restrict__ proj,
                                                 const float* __restrict__ A_log,
                                                 const float* __restrict__ dt_bias,
                                                 float* __restrict__ dtb,
                                                 float* __restrict__ cum) {
  int blk = blockIdx.x;
  int h = blk & 63;
  int l = threadIdx.x;
  int c = (blk >> 6) & 7, b = blk >> 9;
  size_t row = (size_t)(b * SEQ + c * CHUNK + l);
  float raw = proj[row * DP_PAD + OFF_DT + h] + dt_bias[h];
  float sp = raw > 20.f ? raw : log1pf(__expf(raw));
  float dtv = fminf(fmaxf(sp, 1e-4f), 0.5f);
  dtb[row * N_HEADS + h] = dtv;
  float Acoef = -__expf(A_log[h]);
  __shared__ float s[CHUNK];
  s[l] = Acoef * dtv;
  __syncthreads();
  for (int off = 1; off < CHUNK; off <<= 1) {
    float v = (l >= off) ? s[l - off] : 0.f;
    __syncthreads();
    s[l] += v;
    __syncthreads();
  }
  cum[(size_t)blk * CHUNK + l] = s[l];
}

// ---------------------------------------------------------------------------
// delta_h[b,c,h,p,n] = sum_l exp(cum[255]-cum[l]) * x_dt[l,p] * Bc[l,n]
// ---------------------------------------------------------------------------
__global__ __launch_bounds__(256) void delta_h_wmma(
    const float* __restrict__ proj, const float* __restrict__ xcbuf,
    const float* __restrict__ dtb, const float* __restrict__ cum,
    float* __restrict__ dh) {
  __shared__ __bf16 sX[64][136];  // [p][l]
  __shared__ __bf16 sBt[64][136]; // [n][l]
  int blk = blockIdx.x;
  int h = blk & 63, c = (blk >> 6) & 7, b = blk >> 9;
  int tid = threadIdx.x, lane = tid & 31, wv = tid >> 5;
  size_t rowbase = (size_t)(b * SEQ + c * CHUNK);
  const float* cumP = cum + (size_t)blk * CHUNK;
  float cend = cumP[CHUNK - 1];
  v8f acc[2];
  acc[0] = zero8(); acc[1] = zero8();

  for (int half = 0; half < 2; ++half) {
    int l0 = half * 128;
#pragma unroll 4
    for (int it = 0; it < 32; ++it) {
      int idx = tid + it * 256;
      int l = l0 + (idx >> 6);
      int col = idx & 63;
      size_t row = rowbase + l;
      float wgt = __expf(cend - cumP[l]) * dtb[row * N_HEADS + h];
      sX[col][l - l0] = f2bf(xcbuf[row * D_INNER + h * HEAD_DIM + col] * wgt);
      sBt[col][l - l0] = f2bf(proj[row * DP_PAD + OFF_B + h * D_STATE + col]);
    }
    __syncthreads();
#pragma unroll
    for (int tt = 0; tt < 2; ++tt) {
      int tile = wv * 2 + tt;
      int pt = tile >> 2, nt = tile & 3;
#pragma unroll
      for (int kc = 0; kc < 4; ++kc) {
        v16bf ax = afrag_lds(&sX[0][0], 136, pt * 16, kc * 32, lane);
        v16bf bb = bfrag_lds(&sBt[0][0], 136, nt * 16, kc * 32, lane);
        acc[tt] = wmma_bf16(ax, bb, acc[tt]);
      }
    }
    __syncthreads();
  }

  int lo16 = lane & 15, hi8 = (lane >> 4) * 8;
  float* out = dh + (size_t)blk * (HEAD_DIM * D_STATE);
#pragma unroll
  for (int tt = 0; tt < 2; ++tt) {
    int tile = wv * 2 + tt;
    int pt = tile >> 2, nt = tile & 3;
#pragma unroll
    for (int r = 0; r < 8; ++r)
      out[(size_t)(pt * 16 + r + hi8) * D_STATE + nt * 16 + lo16] = acc[tt][r];
  }
}

// ---------------------------------------------------------------------------
// Sequential inter-chunk state recurrence (8 steps).
// ---------------------------------------------------------------------------
__global__ __launch_bounds__(256) void scan_states(const float* __restrict__ dh,
                                                   const float* __restrict__ cum,
                                                   float* __restrict__ st) {
  size_t gid = (size_t)blockIdx.x * 256 + threadIdx.x;
  int n = (int)(gid & 63);
  int p = (int)((gid >> 6) & 63);
  int h = (int)((gid >> 12) & 63);
  int b = (int)(gid >> 18);
  float hs = 0.f;
  for (int c = 0; c < NCHUNK; ++c) {
    size_t blk = (size_t)((b * NCHUNK + c) * 64 + h);
    size_t idx = blk * (HEAD_DIM * D_STATE) + (size_t)p * D_STATE + n;
    st[idx] = hs;
    float cd = __expf(cum[blk * CHUNK + CHUNK - 1]);
    hs = cd * hs + dh[idx];
  }
}

// ---------------------------------------------------------------------------
// Per-(b,c,h) output: y = y_inter + y_intra (WMMA; causal strips).
// ---------------------------------------------------------------------------
__global__ __launch_bounds__(256) void chunk_out_wmma(
    const float* __restrict__ proj, const float* __restrict__ xcbuf,
    const float* __restrict__ dtb, const float* __restrict__ cum,
    const float* __restrict__ st, float* __restrict__ yb) {
  __shared__ __bf16 sS[8][32][40];
  int blk = blockIdx.x;
  int h = blk & 63, c = (blk >> 6) & 7, b = blk >> 9;
  int tid = threadIdx.x, lane = tid & 31, wv = tid >> 5;
  int lo16 = lane & 15, hi8 = (lane >> 4) * 8, hi16 = (lane >> 4) * 16;
  int band = wv * 32;
  size_t rowbase = (size_t)(b * SEQ + c * CHUNK);
  const float* projC = proj + rowbase * DP_PAD + OFF_C + h * D_STATE;
  const float* projB = proj + rowbase * DP_PAD + OFF_B + h * D_STATE;
  const float* cumP = cum + (size_t)blk * CHUNK;
  const float* stP = st + (size_t)blk * (HEAD_DIM * D_STATE);
  const float* xcP = xcbuf + rowbase * D_INNER + h * HEAD_DIM;
  const float* dtP = dtb + rowbase * N_HEADS + h;

  v16bf a_c[2][2];
#pragma unroll
  for (int mt = 0; mt < 2; ++mt)
#pragma unroll
    for (int kc = 0; kc < 2; ++kc)
      a_c[mt][kc] = afrag_g(projC, DP_PAD, band + mt * 16, kc * 32, lane);

  float cum_i[2][8];
#pragma unroll
  for (int mt = 0; mt < 2; ++mt)
#pragma unroll
    for (int r = 0; r < 8; ++r)
      cum_i[mt][r] = cumP[band + mt * 16 + r + hi8];

  v8f acc[2][4];
#pragma unroll
  for (int i = 0; i < 2; ++i)
#pragma unroll
    for (int j = 0; j < 4; ++j) acc[i][j] = zero8();

  // inter-chunk: Cc @ state^T
#pragma unroll
  for (int pt = 0; pt < 4; ++pt) {
    v16bf b0 = bfrag_g(stP, D_STATE, pt * 16, 0, lane);
    v16bf b1 = bfrag_g(stP, D_STATE, pt * 16, 32, lane);
#pragma unroll
    for (int mt = 0; mt < 2; ++mt) {
      acc[mt][pt] = wmma_bf16(a_c[mt][0], b0, acc[mt][pt]);
      acc[mt][pt] = wmma_bf16(a_c[mt][1], b1, acc[mt][pt]);
    }
  }
#pragma unroll
  for (int mt = 0; mt < 2; ++mt) {
    float es[8];
#pragma unroll
    for (int r = 0; r < 8; ++r) es[r] = __expf(cum_i[mt][r]) * SSD_SCALE;
#pragma unroll
    for (int pt = 0; pt < 4; ++pt)
#pragma unroll
      for (int r = 0; r < 8; ++r) acc[mt][pt][r] *= es[r];
  }

  // intra-chunk causal strips
  for (int js = 0; js <= wv; ++js) {
    int jbase = js * 32;
    v16bf b_b[2][2];
#pragma unroll
    for (int jt = 0; jt < 2; ++jt)
#pragma unroll
      for (int kc = 0; kc < 2; ++kc)
        b_b[jt][kc] = bfrag_g(projB, DP_PAD, jbase + jt * 16, kc * 32, lane);

#pragma unroll
    for (int mt = 0; mt < 2; ++mt)
#pragma unroll
      for (int jt = 0; jt < 2; ++jt) {
        v8f s = zero8();
        s = wmma_bf16(a_c[mt][0], b_b[jt][0], s);
        s = wmma_bf16(a_c[mt][1], b_b[jt][1], s);
        int j = jbase + jt * 16 + lo16;
        float cj = cumP[j];
#pragma unroll
        for (int r = 0; r < 8; ++r) {
          int i = band + mt * 16 + r + hi8;
          float v =
              (j <= i) ? __expf(cum_i[mt][r] - cj) * SSD_SCALE * s[r] : 0.f;
          sS[wv][mt * 16 + r + hi8][jt * 16 + lo16] = f2bf(v);
        }
      }

    float dtv[16];
#pragma unroll
    for (int e = 0; e < 16; ++e)
      dtv[e] = dtP[(size_t)(jbase + hi16 + e) * N_HEADS];

    v16bf a_s[2];
#pragma unroll
    for (int mt = 0; mt < 2; ++mt)
      a_s[mt] = afrag_lds(&sS[wv][0][0], 40, mt * 16, 0, lane);

#pragma unroll
    for (int pt = 0; pt < 4; ++pt) {
      v16bf bx;
#pragma unroll
      for (int e = 0; e < 16; ++e) {
        int j = jbase + hi16 + e;
        bx[e] = f2bf(xcP[(size_t)j * D_INNER + pt * 16 + lo16] * dtv[e]);
      }
#pragma unroll
      for (int mt = 0; mt < 2; ++mt)
        acc[mt][pt] = wmma_bf16(a_s[mt], bx, acc[mt][pt]);
    }
  }

#pragma unroll
  for (int mt = 0; mt < 2; ++mt)
#pragma unroll
    for (int pt = 0; pt < 4; ++pt)
#pragma unroll
      for (int r = 0; r < 8; ++r) {
        size_t row = rowbase + band + mt * 16 + r + hi8;
        yb[row * D_INNER + h * HEAD_DIM + pt * 16 + lo16] = acc[mt][pt][r];
      }
}

// ---------------------------------------------------------------------------
// RMS normalization + SiLU(z) gate -> gated (bf16, out_proj GEMM operand).
// ---------------------------------------------------------------------------
__global__ __launch_bounds__(256) void rms_gate(const float* __restrict__ yb,
                                                const float* __restrict__ proj,
                                                __bf16* __restrict__ gated) {
  int row = blockIdx.x;
  int tid = threadIdx.x;
  __shared__ float red[256];
  float ss = 0.f;
  for (int i = tid; i < D_INNER; i += 256) {
    float v = yb[(size_t)row * D_INNER + i];
    ss += v * v;
  }
  red[tid] = ss;
  __syncthreads();
  for (int s = 128; s > 0; s >>= 1) {
    if (tid < s) red[tid] += red[tid + s];
    __syncthreads();
  }
  float inv = rsqrtf(red[0] / (float)D_INNER + RMS_EPS);
  for (int i = tid; i < D_INNER; i += 256) {
    float z = proj[(size_t)row * DP_PAD + i];
    float sil = z / (1.f + __expf(-z));
    gated[(size_t)row * D_INNER + i] =
        f2bf(yb[(size_t)row * D_INNER + i] * inv * sil);
  }
}

// ---------------------------------------------------------------------------
extern "C" void kernel_launch(void* const* d_in, const int* in_sizes, int n_in,
                              void* d_out, int out_size, void* d_ws,
                              size_t ws_size, hipStream_t stream) {
  (void)in_sizes; (void)n_in; (void)out_size; (void)ws_size;
  const float* x          = (const float*)d_in[0];
  const float* in_proj_w  = (const float*)d_in[1];
  const float* conv_w     = (const float*)d_in[2];
  const float* A_log      = (const float*)d_in[3];
  const float* dt_bias    = (const float*)d_in[4];
  const float* out_proj_w = (const float*)d_in[5];
  float* out = (float*)d_out;

  char* w = (char*)d_ws;
  auto carve = [&](size_t bytes) -> void* {
    char* p = w;
    w += (bytes + 255) & ~(size_t)255;
    return p;
  };
  float*  proj   = (float*)carve((size_t)BT * DP_PAD * sizeof(float));
  float*  xcbuf  = (float*)carve((size_t)BT * D_INNER * sizeof(float));
  float*  dtb    = (float*)carve((size_t)BT * N_HEADS * sizeof(float));
  float*  cumb   = (float*)carve((size_t)BATCH * NCHUNK * N_HEADS * CHUNK * sizeof(float));
  float*  dh     = (float*)carve((size_t)BATCH * NCHUNK * N_HEADS * HEAD_DIM * D_STATE * sizeof(float));
  float*  stb    = (float*)carve((size_t)BATCH * NCHUNK * N_HEADS * HEAD_DIM * D_STATE * sizeof(float));
  float*  yb     = (float*)carve((size_t)BT * D_INNER * sizeof(float));
  __bf16* xbf    = (__bf16*)carve((size_t)BT * D_MODEL * sizeof(__bf16));
  __bf16* wbf    = (__bf16*)carve((size_t)DP_PAD * D_MODEL * sizeof(__bf16));
  __bf16* wobf   = (__bf16*)carve((size_t)D_MODEL * D_INNER * sizeof(__bf16));
  __bf16* gbf    = (__bf16*)carve((size_t)BT * D_INNER * sizeof(__bf16));

  // 0) one-time operand conversions to bf16
  cvt_f32_bf16<<<(BT * D_MODEL) / 256, 256, 0, stream>>>(x, xbf);
  cvt_pad_w<<<((size_t)DP_PAD * D_MODEL) / 256, 256, 0, stream>>>(in_proj_w, wbf);
  cvt_f32_bf16<<<(D_MODEL * D_INNER) / 256, 256, 0, stream>>>(out_proj_w, wobf);

  // 1) in_proj: proj[BT, 16448(+pad)] = x @ in_proj_w^T
  gemm_bf16_wmma<DP_PAD, D_MODEL, DP_PAD>
      <<<dim3(DP_PAD / 128, BT / 128), 256, 0, stream>>>(xbf, wbf, proj);
  // 2) depthwise conv + SiLU
  conv_silu<<<(BT * D_INNER) / 256, 256, 0, stream>>>(proj, conv_w, xcbuf);
  // 3) dt + per-chunk cumsum of A*dt
  dt_cumsum<<<BATCH * NCHUNK * N_HEADS, 256, 0, stream>>>(proj, A_log, dt_bias,
                                                          dtb, cumb);
  // 4) per-chunk state contribution (WMMA)
  delta_h_wmma<<<BATCH * NCHUNK * N_HEADS, 256, 0, stream>>>(proj, xcbuf, dtb,
                                                             cumb, dh);
  // 5) inter-chunk recurrence
  scan_states<<<(BATCH * N_HEADS * HEAD_DIM * D_STATE) / 256, 256, 0, stream>>>(
      dh, cumb, stb);
  // 6) intra + inter chunk outputs (WMMA)
  chunk_out_wmma<<<BATCH * NCHUNK * N_HEADS, 256, 0, stream>>>(
      proj, xcbuf, dtb, cumb, stb, yb);
  // 7) RMS norm + gate -> bf16
  rms_gate<<<BT, 256, 0, stream>>>(yb, proj, gbf);
  // 8) out_proj: out[BT, 2048] = gated @ out_proj_w^T
  gemm_bf16_wmma<D_MODEL, D_INNER, D_MODEL>
      <<<dim3(D_MODEL / 128, BT / 128), 256, 0, stream>>>(gbf, wobf, out);
}
